// Encoder_7249904796081
// MI455X (gfx1250) — matrix operand order repaired and checked by reading, not compiled, not run on hardware
//
#include <hip/hip_runtime.h>
#include <hip/hip_bf16.h>
#include <stdint.h>

// Problem constants (match reference)
#define BQ   128
#define TT   1024
#define NNI  256
#define DD   256
#define D3   768

typedef unsigned short ushortT;
typedef __attribute__((ext_vector_type(16))) __bf16 v16bf;
typedef __attribute__((ext_vector_type(8)))  float  v8f;

union Frag { v16bf v; uint4 q[2]; };

__device__ __forceinline__ ushortT f2bf(float f) {
  uint32_t u = __float_as_uint(f);
  u += 0x7fffu + ((u >> 16) & 1u);   // round-to-nearest-even
  return (ushortT)(u >> 16);
}
__device__ __forceinline__ float sigmoidf_(float x) {
  return 1.0f / (1.0f + __expf(-x));
}

// ---------------------------------------------------------------------------
// Kernel: convert data [B,T,N] f32 -> bf16 (vectorized float4 -> 4x bf16)
// ---------------------------------------------------------------------------
__global__ void k_cvt_data(const float* __restrict__ src, ushortT* __restrict__ dst) {
  size_t i = ((size_t)blockIdx.x * blockDim.x + threadIdx.x) * 4;
  float4 f = *(const float4*)(src + i);
  uint32_t lo = (uint32_t)f2bf(f.x) | ((uint32_t)f2bf(f.y) << 16);
  uint32_t hi = (uint32_t)f2bf(f.z) | ((uint32_t)f2bf(f.w) << 16);
  *(uint2*)(dst + i) = make_uint2(lo, hi);
}

// ---------------------------------------------------------------------------
// Kernel: transpose+convert GRU weights.
// src matrices are [K=256][N=768] f32 (row-major). dst is [N=768][K=256] bf16
// so each lane's WMMA B-fragment is two contiguous 8-element K runs.
// blockIdx.y: 0..3 -> Wk of gru g, 4..7 -> U of gru g-4.
// ---------------------------------------------------------------------------
__global__ void k_tw(const float* s0, const float* s1, const float* s2, const float* s3,
                     const float* s4, const float* s5, const float* s6, const float* s7,
                     ushortT* __restrict__ wkt, ushortT* __restrict__ ut) {
  const float* srcs[8] = {s0, s1, s2, s3, s4, s5, s6, s7};
  int m = blockIdx.y;
  const float* src = srcs[m];
  ushortT* dst = (m < 4) ? (wkt + (size_t)m * D3 * DD)
                         : (ut  + (size_t)(m - 4) * D3 * DD);
  int idx = blockIdx.x * blockDim.x + threadIdx.x;   // 0 .. 196607
  int k = idx & 255;
  int n = idx >> 8;                                  // 0..767
  dst[(size_t)n * DD + k] = f2bf(src[(size_t)k * D3 + n]);
}

// ---------------------------------------------------------------------------
// Kernel: zero the CI_LAG padding slots of ci output.
// ci[b][0][0:256] = 0 (fwd pad), ci[b][T-1][256:512] = 0 (bwd pad)
// ---------------------------------------------------------------------------
__global__ void k_zero(float* __restrict__ out_ci) {
  int i = blockIdx.x * blockDim.x + threadIdx.x;     // 0 .. 128*256-1
  int b = i >> 8;
  int c = i & 255;
  out_ci[((size_t)b * TT + 0) * 512 + c] = 0.0f;
  out_ci[((size_t)b * TT + (TT - 1)) * 512 + 256 + c] = 0.0f;
}

// ---------------------------------------------------------------------------
// Main scan kernel. grid = (8 batch-chunks, 4 GRUs), block = 256 (8 waves).
// Each workgroup owns 16 batch rows of one GRU: one M=16 WMMA row-tile.
// Per step:
//   phase 1: zr = sigmoid(b_zr + x_t@Wk[:, :512] + h@U[:, :512])   (32 N-tiles)
//   phase 2: hh = tanh(b_h + x_t@Wk[:,512:] + (r*h)@U[:,512:])     (16 N-tiles)
//   h = clip(z*h + (1-z)*hh, +-5); emit ci sequence / final ic state.
// All matmuls: v_wmma_f32_16x16x32_bf16, f32 accumulation.
// Weights (3 MB bf16 total) are re-read from L2 every step; the opaque
// 32-bit offset below stops LICM from hoisting (and spilling) the weight
// fragment loads out of the time loop WITHOUT destroying the global
// address-space provenance (keeps global_load_b128, not flat_load).
// ---------------------------------------------------------------------------
__launch_bounds__(256)
__global__ void k_scan(const ushortT* __restrict__ dataBF,
                       const ushortT* __restrict__ wkt,
                       const ushortT* __restrict__ ut,
                       const float* __restrict__ b_ic_f, const float* __restrict__ b_ic_b,
                       const float* __restrict__ b_ci_f, const float* __restrict__ b_ci_b,
                       const float* __restrict__ ic_h0, const float* __restrict__ ci_h0,
                       float* __restrict__ out_ci,
                       float* __restrict__ hlast) {
  __shared__ __align__(16) float   h_f[16 * DD];   // f32 hidden state
  __shared__ __align__(16) float   z_f[16 * DD];   // z gate
  __shared__ __align__(16) ushortT h_b[16 * DD];   // bf16 hidden (A operand)
  __shared__ __align__(16) ushortT a_b[16 * DD];   // bf16 r*h (A operand)

  const int  g        = blockIdx.y;
  const int  m0       = blockIdx.x * 16;
  const bool backward = (g & 1);
  const bool is_ci    = (g >= 2);
  const ushortT* Wt_base = wkt + (size_t)g * D3 * DD;
  const ushortT* Ut_base = ut  + (size_t)g * D3 * DD;
  const float* bias = (g == 0) ? b_ic_f : (g == 1) ? b_ic_b : (g == 2) ? b_ci_f : b_ci_b;
  const float* h0   = ((g < 2) ? ic_h0 : ci_h0) + (g & 1) * DD;

  const int tid  = threadIdx.x;
  const int lane = tid & 31;
  const int wave = tid >> 5;

  // Init hidden state (h0 row broadcast over the 16 batch rows)
  for (int i = tid; i < 16 * DD; i += 256) {
    float v = h0[i & 255];
    h_f[i] = v;
    h_b[i] = f2bf(v);
  }
  __syncthreads();

  // WMMA lane decomposition (16x16x32 bf16, wave32):
  //  A: lane holds row m=lane&15, K in {s..s+7} u {s+16..s+23}, s=8*(lane>=16)
  //  B: lane holds col n=lane&15, K in {s..s+7} u {s+16..s+23} (same as A)
  //  C/D: lane holds col n=lane&15, rows m = e + 8*(lane>=16), e=0..7
  const int am   = lane & 15;
  const int s8   = (lane >> 4) * 8;
  const int s16  = (lane >> 4) * 16;
  const int crow = (lane >> 4) * 8;

  const ushortT* xrow_base = dataBF + (size_t)(m0 + am) * TT * NNI;

  for (int t = 0; t < TT; ++t) {
    const int t_x = backward ? (TT - 1 - t) : t;
    const ushortT* xk = xrow_base + (size_t)t_x * NNI;

    // Defeat LICM without losing addrspace(1) provenance: opaque ZERO offset
    // (uniform SGPR) added to the weight bases each iteration. Loads stay in
    // the loop (hit L2 every step) and remain global_load_b128 with
    // SGPR-base + 32-bit voffset addressing.
    unsigned zoff = 0;
    asm volatile("" : "+s"(zoff));
    const ushortT* Wt = Wt_base + zoff;
    const ushortT* Ut = Ut_base + zoff;

    // Prefetch next step's x row into cache (gfx1250 global_prefetch_b8)
    if (t + 1 < TT) {
      const int t_n = backward ? (TT - 2 - t) : (t + 1);
      __builtin_prefetch(xrow_base + (size_t)t_n * NNI, 0, 1);
    }

    // -------- phase 1: zr pre-activations (512 cols; 4 N-tiles per wave) ----
    v8f acc[4] = {};            // bias folded into the epilogue
    int n0[4];
#pragma unroll
    for (int j = 0; j < 4; ++j) n0[j] = (wave + 8 * j) * 16;

    // x_t contribution (K = 0..255 against Wk[:, :512])
#pragma unroll 2
    for (int k0 = 0; k0 < 8; ++k0) {
      Frag a;
      a.q[0] = *(const uint4*)(xk + k0 * 32 + s8);
      a.q[1] = *(const uint4*)(xk + k0 * 32 + s8 + 16);
#pragma unroll
      for (int j = 0; j < 4; ++j) {
        Frag bf;
        const ushortT* bp = Wt + (size_t)(n0[j] + am) * DD + k0 * 32 + s8;
        bf.q[0] = *(const uint4*)(bp);
        bf.q[1] = *(const uint4*)(bp + 16);
        acc[j] = __builtin_amdgcn_wmma_f32_16x16x32_bf16(
            false, a.v, false, bf.v, (short)0, acc[j], false, false);
      }
    }
    // h contribution (K = 0..255 against U[:, :512]); A from LDS
#pragma unroll 2
    for (int k0 = 0; k0 < 8; ++k0) {
      Frag a;
      const ushortT* hp = h_b + am * DD + k0 * 32 + s8;
      a.q[0] = *(const uint4*)(hp);
      a.q[1] = *(const uint4*)(hp + 16);
#pragma unroll
      for (int j = 0; j < 4; ++j) {
        Frag bf;
        const ushortT* bp = Ut + (size_t)(n0[j] + am) * DD + k0 * 32 + s8;
        bf.q[0] = *(const uint4*)(bp);
        bf.q[1] = *(const uint4*)(bp + 16);
        acc[j] = __builtin_amdgcn_wmma_f32_16x16x32_bf16(
            false, a.v, false, bf.v, (short)0, acc[j], false, false);
      }
    }
    // gates -> LDS: z (f32) and a = bf16(r * h); bias added here
#pragma unroll
    for (int j = 0; j < 4; ++j) {
      int n = n0[j] + am;
      float bn = bias[n];
#pragma unroll
      for (int e = 0; e < 8; ++e) {
        int m = crow + e;
        float s = sigmoidf_(acc[j][e] + bn);
        if (n < 256) {
          z_f[m * DD + n] = s;
        } else {
          int nn = n - 256;
          a_b[m * DD + nn] = f2bf(s * h_f[m * DD + nn]);
        }
      }
    }
    __syncthreads();

    // -------- phase 2: candidate (256 cols; 2 N-tiles per wave) -------------
    v8f acc2[2] = {};
    int nn0[2];
#pragma unroll
    for (int j = 0; j < 2; ++j) nn0[j] = (wave + 8 * j) * 16;

#pragma unroll 2
    for (int k0 = 0; k0 < 8; ++k0) {
      Frag a;
      a.q[0] = *(const uint4*)(xk + k0 * 32 + s8);
      a.q[1] = *(const uint4*)(xk + k0 * 32 + s8 + 16);
#pragma unroll
      for (int j = 0; j < 2; ++j) {
        Frag bf;
        const ushortT* bp = Wt + (size_t)(512 + nn0[j] + am) * DD + k0 * 32 + s8;
        bf.q[0] = *(const uint4*)(bp);
        bf.q[1] = *(const uint4*)(bp + 16);
        acc2[j] = __builtin_amdgcn_wmma_f32_16x16x32_bf16(
            false, a.v, false, bf.v, (short)0, acc2[j], false, false);
      }
    }
#pragma unroll 2
    for (int k0 = 0; k0 < 8; ++k0) {
      Frag a;
      const ushortT* ap = a_b + am * DD + k0 * 32 + s8;
      a.q[0] = *(const uint4*)(ap);
      a.q[1] = *(const uint4*)(ap + 16);
#pragma unroll
      for (int j = 0; j < 2; ++j) {
        Frag bf;
        const ushortT* bp = Ut + (size_t)(512 + nn0[j] + am) * DD + k0 * 32 + s8;
        bf.q[0] = *(const uint4*)(bp);
        bf.q[1] = *(const uint4*)(bp + 16);
        acc2[j] = __builtin_amdgcn_wmma_f32_16x16x32_bf16(
            false, a.v, false, bf.v, (short)0, acc2[j], false, false);
      }
    }
    // h update + outputs (ci stream stored non-temporally: 256 MB > 192 MB L2)
#pragma unroll
    for (int j = 0; j < 2; ++j) {
      int nn = nn0[j] + am;
      float bn = bias[512 + nn];
#pragma unroll
      for (int e = 0; e < 8; ++e) {
        int m = crow + e;
        float hh = tanhf(acc2[j][e] + bn);
        float zv = z_f[m * DD + nn];
        float ho = h_f[m * DD + nn];
        float hn = zv * ho + (1.0f - zv) * hh;
        hn = fminf(5.0f, fmaxf(-5.0f, hn));
        h_f[m * DD + nn] = hn;
        h_b[m * DD + nn] = f2bf(hn);
        if (is_ci) {
          if (t < TT - 1) {
            int b = m0 + m;
            if (!backward)
              __builtin_nontemporal_store(hn,
                  &out_ci[((size_t)b * TT + (t + 1)) * 512 + nn]);            // ci_fwd lagged
            else
              __builtin_nontemporal_store(hn,
                  &out_ci[((size_t)b * TT + (TT - 2 - t)) * 512 + 256 + nn]); // ci_bwd lagged
          }
        } else if (t == TT - 1) {
          hlast[(size_t)g * BQ * DD + (size_t)(m0 + m) * DD + nn] = hn;
        }
      }
    }
    __syncthreads();
  }
}

// ---------------------------------------------------------------------------
// Heads: ic_mean / ic_stddev from concat(hf, hb) [128, 512]. Tiny (33 MFLOP).
// ---------------------------------------------------------------------------
__global__ void k_proj(const float* __restrict__ hlast,
                       const float* __restrict__ Wm, const float* __restrict__ bm,
                       const float* __restrict__ Wl, const float* __restrict__ bl,
                       float* __restrict__ out) {
  int i = blockIdx.x * blockDim.x + threadIdx.x;   // 0 .. 16383
  int b = i >> 7;
  int j = i & 127;
  const float* hf = hlast + (size_t)b * DD;                   // ic fwd final state
  const float* hb = hlast + (size_t)BQ * DD + (size_t)b * DD; // ic bwd final state
  float m = bm[j], l = bl[j];
  for (int k = 0; k < 256; ++k) {
    float h = hf[k];
    m = fmaf(h, Wm[(size_t)k * 128 + j], m);
    l = fmaf(h, Wl[(size_t)k * 128 + j], l);
  }
  for (int k = 0; k < 256; ++k) {
    float h = hb[k];
    m = fmaf(h, Wm[(size_t)(256 + k) * 128 + j], m);
    l = fmaf(h, Wl[(size_t)(256 + k) * 128 + j], l);
  }
  out[i] = m;                                    // ic_mean
  out[128 * 128 + i] = sqrtf(__expf(l) + 1e-4f); // ic_stddev
}

// ---------------------------------------------------------------------------
// Host-side launcher
// ---------------------------------------------------------------------------
extern "C" void kernel_launch(void* const* d_in, const int* in_sizes, int n_in,
                              void* d_out, int out_size, void* d_ws, size_t ws_size,
                              hipStream_t stream) {
  (void)in_sizes; (void)n_in; (void)out_size; (void)ws_size;

  const float* data = (const float*)d_in[0];

  // Workspace layout
  uint8_t* ws = (uint8_t*)d_ws;
  const size_t DATA_BF_BYTES = (size_t)BQ * TT * NNI * 2;        // 64 MB
  const size_t WMAT_BYTES    = (size_t)4 * D3 * DD * 2;          // 1.5 MB each
  ushortT* dataBF = (ushortT*)ws;
  ushortT* wkt    = (ushortT*)(ws + DATA_BF_BYTES);
  ushortT* ut     = (ushortT*)(ws + DATA_BF_BYTES + WMAT_BYTES);
  float*   hlast  = (float*)(ws + DATA_BF_BYTES + 2 * WMAT_BYTES); // 2*128*256 f32

  float* out    = (float*)d_out;
  float* out_ci = out + 2 * 128 * 128;

  // 1) data f32 -> bf16
  k_cvt_data<<<(BQ * TT * NNI) / 4 / 256, 256, 0, stream>>>(data, dataBF);

  // 2) transpose + convert all 8 weight matrices (Wk x4, U x4)
  dim3 gtw(768, 8);
  k_tw<<<gtw, 256, 0, stream>>>(
      (const float*)d_in[1], (const float*)d_in[4],
      (const float*)d_in[8], (const float*)d_in[11],
      (const float*)d_in[2], (const float*)d_in[5],
      (const float*)d_in[9], (const float*)d_in[12],
      wkt, ut);

  // 3) zero the ci lag-padding slots
  k_zero<<<128, 256, 0, stream>>>(out_ci);

  // 4) four GRU scans, 8 batch-chunks each, fully independent workgroups
  dim3 gscan(8, 4);
  k_scan<<<gscan, 256, 0, stream>>>(
      dataBF, wkt, ut,
      (const float*)d_in[3], (const float*)d_in[6],
      (const float*)d_in[10], (const float*)d_in[13],
      (const float*)d_in[7], (const float*)d_in[14],
      out_ci, hlast);

  // 5) ic_mean / ic_stddev heads
  k_proj<<<64, 256, 0, stream>>>(
      hlast,
      (const float*)d_in[15], (const float*)d_in[16],
      (const float*)d_in[17], (const float*)d_in[18],
      out);
}
